// DomainProjectionLDP_12455405158618
// MI455X (gfx1250) — matrix operand order repaired and checked by reading, not compiled, not run on hardware
//
#include <hip/hip_runtime.h>

#define B_ROWS 8192
#define D_DIM  1024
#define N_DOM  8
#define TILES_MAX (B_ROWS / 16 + N_DOM)   // 520 upper bound on row tiles
#define REG_BLOCKS 512

typedef __attribute__((ext_vector_type(16))) __bf16 bf16x16;
typedef __attribute__((ext_vector_type(8)))  __bf16 bf16x8;
typedef __attribute__((ext_vector_type(8)))  float  f32x8;

// ---------------------------------------------------------------------------
// Split an fp32 value into hi/lo bf16 parts: x ~= hi + lo (error ~2^-16 rel)
// ---------------------------------------------------------------------------
__device__ __forceinline__ void split_bf16(float x, __bf16& h, __bf16& l) {
    unsigned u = __builtin_bit_cast(unsigned, x);
    h = __builtin_bit_cast(__bf16, (unsigned short)(u >> 16));
    float hf = __builtin_bit_cast(float, u & 0xFFFF0000u);
    float r  = x - hf;
    unsigned v = __builtin_bit_cast(unsigned, r);
    l = __builtin_bit_cast(__bf16, (unsigned short)(v >> 16));
}

// ---------------------------------------------------------------------------
// One-time pre-pass: split an f32 array into hi/lo bf16 arrays (row-major).
// 8 floats per thread -> one 16B hi store + one 16B lo store.
// ---------------------------------------------------------------------------
__global__ __launch_bounds__(256) void ldp_split(const float* __restrict__ src,
                                                 __bf16* __restrict__ hi,
                                                 __bf16* __restrict__ lo,
                                                 int n8) {
    int i = blockIdx.x * 256 + threadIdx.x;
    if (i >= n8) return;
    const float4* p = (const float4*)src + 2 * (size_t)i;
    float4 f0 = p[0], f1 = p[1];
    float fv[8] = {f0.x, f0.y, f0.z, f0.w, f1.x, f1.y, f1.z, f1.w};
    bf16x8 h, l;
#pragma unroll
    for (int j = 0; j < 8; ++j) {
        __bf16 hh, ll;
        split_bf16(fv[j], hh, ll);
        h[j] = hh; l[j] = ll;
    }
    *((bf16x8*)hi + i) = h;
    *((bf16x8*)lo + i) = l;
}

// ---------------------------------------------------------------------------
// ws layout: [0..8) counts, [8..16) offsets, [16..24) cursors (ints)
//            [32..32+8192) rowIdx (ints); partials floats after that;
//            64KB: muH(16MB) muL(16MB) WH(16MB) WL(16MB)
// ---------------------------------------------------------------------------
__global__ void ldp_init(int* ws) {
    int t = threadIdx.x;
    if (t < 24) ws[t] = 0;
}

__global__ void ldp_hist(const int* __restrict__ ids, int* __restrict__ counts) {
    int b = blockIdx.x * blockDim.x + threadIdx.x;
    if (b < B_ROWS) atomicAdd(&counts[ids[b]], 1);
}

__global__ void ldp_scan(const int* __restrict__ counts, int* __restrict__ offsets) {
    if (threadIdx.x == 0) {
        int acc = 0;
        for (int n = 0; n < N_DOM; ++n) { offsets[n] = acc; acc += counts[n]; }
    }
}

__global__ void ldp_scatter(const int* __restrict__ ids, const int* __restrict__ offsets,
                            int* __restrict__ cursors, int* __restrict__ rowIdx) {
    int b = blockIdx.x * blockDim.x + threadIdx.x;
    if (b < B_ROWS) {
        int id  = ids[b];
        int pos = offsets[id] + atomicAdd(&cursors[id], 1);
        rowIdx[pos] = b;
    }
}

// ---------------------------------------------------------------------------
// Grouped GEMM on pre-split bf16 data: out[row,:] = mu[row,:] @ W[dom]^T.
// Block = 256 threads = 8 waves; each wave owns a 16x16 tile; block covers
// 16 rows x 128 cols. grid = (D/128, TILES_MAX). Inner loop: 8 b128 loads +
// 3 chained v_wmma_f32_16x16x32_bf16 (hi*hi + hi*lo + lo*hi).
// ---------------------------------------------------------------------------
__global__ __launch_bounds__(256) void ldp_gemm(const __bf16* __restrict__ muH,
                                                const __bf16* __restrict__ muL,
                                                const __bf16* __restrict__ WH,
                                                const __bf16* __restrict__ WL,
                                                const int*    __restrict__ rowIdx,
                                                const int*    __restrict__ counts,
                                                const int*    __restrict__ offsets,
                                                float*        __restrict__ out) {
    const int t = blockIdx.y;
    int dom = -1, tin = 0, cnt = 0, off = 0, acc = 0;
#pragma unroll
    for (int n = 0; n < N_DOM; ++n) {
        int c  = counts[n];
        int nt = (c + 15) >> 4;
        if (dom < 0 && t < acc + nt) { dom = n; tin = t - acc; cnt = c; off = offsets[n]; }
        acc += nt;
    }
    if (dom < 0) return;   // uniform per block: EXEC all-ones for WMMA

    const int lane = threadIdx.x & 31;
    const int wave = threadIdx.x >> 5;
    const int half = lane >> 4;
    const int m    = lane & 15;
    const int col  = blockIdx.x * 128 + wave * 16 + m;

    // A-fragment row (lanes 0-15 and 16-31 both carry M = lane&15)
    const int gA   = tin * 16 + m;
    const int rowA = rowIdx[off + (gA < cnt ? gA : 0)];
    const __bf16* __restrict__ pAh = muH + (size_t)rowA * D_DIM;
    const __bf16* __restrict__ pAl = muL + (size_t)rowA * D_DIM;
    const __bf16* __restrict__ pBh = WH + (size_t)dom * D_DIM * D_DIM + (size_t)col * D_DIM;
    const __bf16* __restrict__ pBl = WL + (size_t)dom * D_DIM * D_DIM + (size_t)col * D_DIM;

    // Hoist loop-invariant store rows / masks
    int  rowS[8];
    bool valS[8];
#pragma unroll
    for (int v = 0; v < 8; ++v) {
        int M = v + (half << 3);
        int g = tin * 16 + M;
        valS[v] = (g < cnt);
        rowS[v] = rowIdx[off + (valS[v] ? g : 0)];
    }

    const int koff8  = half << 3;   // A: K = d + koff8 + {0..7} and +16
    const int koff16 = half << 4;   // unused; B also uses K = d + koff8 + {0..7} and +16

    f32x8 c = {0.f, 0.f, 0.f, 0.f, 0.f, 0.f, 0.f, 0.f};

#pragma unroll 2
    for (int d = 0; d < D_DIM; d += 32) {
        const __bf16* a0 = pAh + d + koff8;
        const __bf16* a1 = pAl + d + koff8;
        const __bf16* b0 = pBh + d + koff8;
        const __bf16* b1 = pBl + d + koff8;

        bf16x8 ah0 = *(const bf16x8*)(a0);
        bf16x8 ah1 = *(const bf16x8*)(a0 + 16);
        bf16x8 al0 = *(const bf16x8*)(a1);
        bf16x8 al1 = *(const bf16x8*)(a1 + 16);
        bf16x8 bh0 = *(const bf16x8*)(b0);
        bf16x8 bh1 = *(const bf16x8*)(b0 + 16);
        bf16x8 bl0 = *(const bf16x8*)(b1);
        bf16x8 bl1 = *(const bf16x8*)(b1 + 16);

        __builtin_prefetch(b0 + 32, 0, 0);   // global_prefetch_b8, next K-slab
        __builtin_prefetch(a0 + 32, 0, 0);

        bf16x16 ah = __builtin_shufflevector(ah0, ah1, 0,1,2,3,4,5,6,7,8,9,10,11,12,13,14,15);
        bf16x16 al = __builtin_shufflevector(al0, al1, 0,1,2,3,4,5,6,7,8,9,10,11,12,13,14,15);
        bf16x16 bh = __builtin_shufflevector(bh0, bh1, 0,1,2,3,4,5,6,7,8,9,10,11,12,13,14,15);
        bf16x16 bl = __builtin_shufflevector(bl0, bl1, 0,1,2,3,4,5,6,7,8,9,10,11,12,13,14,15);

        c = __builtin_amdgcn_wmma_f32_16x16x32_bf16(false, ah, false, bh, (short)0, c, false, false);
        c = __builtin_amdgcn_wmma_f32_16x16x32_bf16(false, ah, false, bl, (short)0, c, false, false);
        c = __builtin_amdgcn_wmma_f32_16x16x32_bf16(false, al, false, bh, (short)0, c, false, false);
    }

    // C/D layout: lane holds column `col`; VGPR v holds row M = v + half*8
#pragma unroll
    for (int v = 0; v < 8; ++v) {
        if (valS[v]) out[(size_t)rowS[v] * D_DIM + col] = c[v];
    }
}

// ---------------------------------------------------------------------------
// reg loss: sum over (k,d) of (sum_n W^2 - (sum_n W)^2/N), then /(N*D*D).
// ---------------------------------------------------------------------------
__global__ __launch_bounds__(256) void ldp_reg1(const float* __restrict__ W,
                                                float* __restrict__ partials) {
    const int P = D_DIM * D_DIM;
    const int stride = REG_BLOCKS * 256;
    float local = 0.f;
    for (int p = blockIdx.x * 256 + threadIdx.x; p < P; p += stride) {
        float s1 = 0.f, s2 = 0.f;
#pragma unroll
        for (int n = 0; n < N_DOM; ++n) {
            float w = W[(size_t)n * P + p];
            s1 += w;
            s2 += w * w;
        }
        local += s2 - s1 * s1 * (1.0f / N_DOM);
    }
    __shared__ float sm[256];
    sm[threadIdx.x] = local;
    __syncthreads();
    for (int s = 128; s > 0; s >>= 1) {
        if (threadIdx.x < s) sm[threadIdx.x] += sm[threadIdx.x + s];
        __syncthreads();
    }
    if (threadIdx.x == 0) partials[blockIdx.x] = sm[0];
}

__global__ __launch_bounds__(256) void ldp_reg2(const float* __restrict__ partials,
                                                float* __restrict__ out_scalar) {
    __shared__ float sm[256];
    float v = partials[threadIdx.x] + partials[threadIdx.x + 256];
    sm[threadIdx.x] = v;
    __syncthreads();
    for (int s = 128; s > 0; s >>= 1) {
        if (threadIdx.x < s) sm[threadIdx.x] += sm[threadIdx.x + s];
        __syncthreads();
    }
    if (threadIdx.x == 0)
        out_scalar[0] = sm[0] * (1.0f / ((float)N_DOM * (float)D_DIM * (float)D_DIM));
}

// ---------------------------------------------------------------------------
extern "C" void kernel_launch(void* const* d_in, const int* in_sizes, int n_in,
                              void* d_out, int out_size, void* d_ws, size_t ws_size,
                              hipStream_t stream) {
    (void)in_sizes; (void)n_in; (void)out_size; (void)ws_size;
    const float* mu  = (const float*)d_in[0];
    const int*   ids = (const int*)d_in[1];
    const float* W   = (const float*)d_in[2];
    float* out = (float*)d_out;

    int* wsI        = (int*)d_ws;
    int* counts     = wsI;
    int* offsets    = wsI + 8;
    int* cursors    = wsI + 16;
    int* rowIdx     = wsI + 32;
    float* partials = (float*)(wsI + 32 + B_ROWS);

    const size_t MU_E = (size_t)B_ROWS * D_DIM;          // 8M elements
    const size_t W_E  = (size_t)N_DOM * D_DIM * D_DIM;   // 8M elements
    char* big = (char*)d_ws + (64 << 10);
    __bf16* muH = (__bf16*)big;
    __bf16* muL = muH + MU_E;
    __bf16* WH  = muL + MU_E;
    __bf16* WL  = WH + W_E;

    ldp_init   <<<1, 32, 0, stream>>>(wsI);
    ldp_hist   <<<B_ROWS / 256, 256, 0, stream>>>(ids, counts);
    ldp_scan   <<<1, 32, 0, stream>>>(counts, offsets);
    ldp_scatter<<<B_ROWS / 256, 256, 0, stream>>>(ids, offsets, cursors, rowIdx);

    const int muN8 = (int)(MU_E / 8);
    const int wN8  = (int)(W_E / 8);
    ldp_split<<<(muN8 + 255) / 256, 256, 0, stream>>>(mu, muH, muL, muN8);
    ldp_split<<<(wN8 + 255) / 256, 256, 0, stream>>>(W, WH, WL, wN8);

    dim3 grid(D_DIM / 128, TILES_MAX);
    ldp_gemm<<<grid, 256, 0, stream>>>(muH, muL, WH, WL, rowIdx, counts, offsets, out);

    ldp_reg1<<<REG_BLOCKS, 256, 0, stream>>>(W, partials);
    ldp_reg2<<<1, 256, 0, stream>>>(partials, out + (size_t)B_ROWS * D_DIM);
}
